// EdgeModelSimple_88587995447604
// MI455X (gfx1250) — matrix operand order repaired and checked by reading, not compiled, not run on hardware
//
#include <hip/hip_runtime.h>
#include <hip/hip_bf16.h>

// ---- problem constants (match reference) ----
#define BN 8
#define NN 1024
#define EE 4096
#define FSZ 256
#define CHN 32
#define DV 7
#define VOX (DV * DV * DV)

typedef __attribute__((ext_vector_type(16))) __bf16         v16bf;
typedef __attribute__((ext_vector_type(2)))  __bf16         v2bf;
typedef __attribute__((ext_vector_type(8)))  float          v8f;
typedef __attribute__((ext_vector_type(8)))  unsigned short v8u;
typedef __attribute__((ext_vector_type(16))) unsigned short v16u;

union U16 { v16u u; v16bf b; v8u h[2]; };

__device__ __forceinline__ unsigned short f2bf(float f) {
  unsigned int u = __float_as_uint(f);
  u += 0x7FFFu + ((u >> 16) & 1u);          // round-to-nearest-even
  return (unsigned short)(u >> 16);
}
__device__ __forceinline__ float bf2f(unsigned short s) {
  return __uint_as_float(((unsigned int)s) << 16);
}
// packed f32x2 -> bf16x2 (low = a, high = b)
#if __has_builtin(__builtin_amdgcn_cvt_pk_bf16_f32)
__device__ __forceinline__ unsigned int f2bf_pk(float a, float b) {
  union { v2bf v; unsigned int u; } c;
  c.v = __builtin_amdgcn_cvt_pk_bf16_f32(a, b);
  return c.u;
}
#else
__device__ __forceinline__ unsigned int f2bf_pk(float a, float b) {
  return (unsigned int)f2bf(a) | ((unsigned int)f2bf(b) << 16);
}
#endif

// ---------------- generic WMMA GEMM ----------------
// Out[M,N] = act( sum_segs( A_seg[gather] @ Bsw ) + bias )
// A: row-major bf16 (AF32=false) or f32 staged via LDS (AF32=true).
// Bsw swizzled bf16: element (k,n) at ktile*(N*32) + n*32 + (k&31).
// K per seg multiple of 32, M multiple of 128, N multiple of 128 (always 256).
struct Seg {
  const void* A;
  const int*  gather;   // global row indices, or nullptr
  long        aBatch;   // per-batch element stride of A
  int         lda;      // leading dim (elements)
  int         K;        // multiple of 32
};
struct GemmP {
  Seg  seg[3];
  int  nseg;
  const unsigned short* Bsw;
  long bBatch;
  const float* bias;    // nullptr ok
  unsigned short* Out;  // bf16 row-major
  int  ldo;
  long oBatch;
  int  M, N;
  int  act;             // 0 = none, 1 = ELU
};

template <bool AF32>
__global__ __launch_bounds__(256) void gemm_bf16_t(GemmP p) {
  const int tid  = threadIdx.x;
  const int lane = tid & 31;
  const int wave = tid >> 5;
  const int wm = wave & 3, wn = wave >> 2;
  const int hi = lane >> 4, ml = lane & 15;
  const int mBase = blockIdx.x * 128;
  const int m0 = mBase + wm * 32;
  const int n0 = blockIdx.y * 128 + wn * 64;
  const int b  = blockIdx.z;
  const long bswStride = (long)p.N * 32;

  // LDS staging for f32 A (adj): 128 rows x 32 k of bf16, double buffered
  __shared__ unsigned short lta[2][128 * 32];

  v8f acc[2][4] = {};

  int ktg = 0;
  for (int s = 0; s < p.nseg; ++s) {
    Seg sg = p.seg[s];
    int row[2];
#pragma unroll
    for (int mt = 0; mt < 2; ++mt) {
      int r = m0 + mt * 16 + ml;
      row[mt] = sg.gather ? sg.gather[r] : r;
    }
    // staging thread's source row (AF32 path)
    int srow = 0;
    if (AF32) {
      int lr = tid >> 1;
      srow = sg.gather ? sg.gather[mBase + lr] : (mBase + lr);
    }
    const int nkt = sg.K >> 5;
    for (int kt = 0; kt < nkt; ++kt, ++ktg) {
      const int k0 = kt << 5;
      U16 afr[2];
      if (AF32) {
        const int buf = ktg & 1;
        // cooperative stage: each thread converts 16 floats of one row-half
        {
          const int lr = tid >> 1, half = tid & 1;
          const float* ap = (const float*)sg.A + (long)b * sg.aBatch +
                            (long)srow * sg.lda + k0 + half * 16;
          v8f f0 = *(const v8f*)(ap);
          v8f f1 = *(const v8f*)(ap + 8);
          v8u o0, o1;
#pragma unroll
          for (int j = 0; j < 4; ++j) {
            unsigned int pk0 = f2bf_pk(f0[2 * j], f0[2 * j + 1]);
            unsigned int pk1 = f2bf_pk(f1[2 * j], f1[2 * j + 1]);
            o0[2 * j] = (unsigned short)pk0; o0[2 * j + 1] = (unsigned short)(pk0 >> 16);
            o1[2 * j] = (unsigned short)pk1; o1[2 * j + 1] = (unsigned short)(pk1 >> 16);
          }
          unsigned short* dst = &lta[buf][lr * 32 + half * 16];
          *(v8u*)(dst)     = o0;
          *(v8u*)(dst + 8) = o1;
        }
        __syncthreads();
#pragma unroll
        for (int mt = 0; mt < 2; ++mt) {
          const unsigned short* ap = &lta[buf][(wm * 32 + mt * 16 + ml) * 32];
          afr[mt].h[0] = *(const v8u*)(ap + hi * 8);
          afr[mt].h[1] = *(const v8u*)(ap + 16 + hi * 8);
        }
      } else {
        const unsigned short* Au = (const unsigned short*)sg.A + (long)b * sg.aBatch;
#pragma unroll
        for (int mt = 0; mt < 2; ++mt) {
          const unsigned short* ap = Au + (long)row[mt] * sg.lda + k0;
          afr[mt].h[0] = *(const v8u*)(ap + hi * 8);       // K = hi*8 .. +7
          afr[mt].h[1] = *(const v8u*)(ap + 16 + hi * 8);  // K = 16+hi*8 .. +7
        }
      }
      // B fragments: lane holds col n = ml (+tile), K = hi*8 .. +7 and 16+hi*8 .. +7
      const unsigned short* bp =
          p.Bsw + (long)b * p.bBatch + (long)ktg * bswStride;
      U16 bfr[4];
#pragma unroll
      for (int nt = 0; nt < 4; ++nt) {
        const unsigned short* q = bp + (long)(n0 + nt * 16 + ml) * 32;
        bfr[nt].h[0] = *(const v8u*)(q + hi * 8);
        bfr[nt].h[1] = *(const v8u*)(q + 16 + hi * 8);
      }
#pragma unroll
      for (int mt = 0; mt < 2; ++mt)
#pragma unroll
        for (int nt = 0; nt < 4; ++nt)
          acc[mt][nt] = __builtin_amdgcn_wmma_f32_16x16x32_bf16(
              false, afr[mt].b, false, bfr[nt].b, (short)0, acc[mt][nt],
              false, false);
    }
  }

  // epilogue: bias + (ELU) + bf16 store, C layout: m = r + 8*hi, n = ml
  unsigned short* outB = p.Out + (long)b * p.oBatch;
#pragma unroll
  for (int nt = 0; nt < 4; ++nt) {
    const int n = n0 + nt * 16 + ml;
    const float bv = p.bias ? p.bias[n] : 0.f;
#pragma unroll
    for (int mt = 0; mt < 2; ++mt) {
      const int mrow = m0 + mt * 16 + hi * 8;
#pragma unroll
      for (int r = 0; r < 8; r += 2) {
        float x0 = acc[mt][nt][r]     + bv;
        float x1 = acc[mt][nt][r + 1] + bv;
        if (p.act) {
          x0 = x0 > 0.f ? x0 : expm1f(x0);
          x1 = x1 > 0.f ? x1 : expm1f(x1);
        }
        unsigned int pk = f2bf_pk(x0, x1);
        outB[(long)(mrow + r)     * p.ldo + n] = (unsigned short)pk;
        outB[(long)(mrow + r + 1) * p.ldo + n] = (unsigned short)(pk >> 16);
      }
    }
  }
}

// ---------------- helper kernels ----------------
__global__ __launch_bounds__(1024) void minmax_kernel(const float* __restrict__ v,
                                                      int n, float* __restrict__ out) {
  __shared__ float smin[1024], smax[1024];
  float mn = 3.0e38f, mx = -3.0e38f;
  for (int i = threadIdx.x; i < n; i += 1024) {
    float x = v[i];
    mn = fminf(mn, x);
    mx = fmaxf(mx, x);
  }
  smin[threadIdx.x] = mn; smax[threadIdx.x] = mx;
  __syncthreads();
  for (int s = 512; s > 0; s >>= 1) {
    if ((int)threadIdx.x < s) {
      smin[threadIdx.x] = fminf(smin[threadIdx.x], smin[threadIdx.x + s]);
      smax[threadIdx.x] = fmaxf(smax[threadIdx.x], smax[threadIdx.x + s]);
    }
    __syncthreads();
  }
  if (threadIdx.x == 0) { out[0] = smin[0]; out[1] = smax[0]; }
}

__global__ void gidx_kernel(const int* __restrict__ v1, const int* __restrict__ v2,
                            int* __restrict__ g1, int* __restrict__ g2) {
  int i = blockIdx.x * blockDim.x + threadIdx.x;
  if (i >= BN * EE) return;
  int b = i / EE;
  g1[i] = b * NN + v1[i];
  g2[i] = b * NN + v2[i];
}

// f32 weight [K,N] -> zero-padded, B-swizzled bf16
__global__ void swizw_kernel(const float* __restrict__ W, unsigned short* __restrict__ out,
                             int K, int Kpad, int N) {
  int i = blockIdx.x * blockDim.x + threadIdx.x;
  if (i >= Kpad * N) return;
  int k = i / N, n = i % N;
  float v = (k < K) ? W[(long)k * N + n] : 0.f;
  out[(long)(k >> 5) * (N * 32) + n * 32 + (k & 31)] = f2bf(v);
}

// stage1_out bf16 [B,E,F] row-major -> per-batch B-swizzled (K = e)
__global__ void swizb_kernel(const unsigned short* __restrict__ in,
                             unsigned short* __restrict__ out) {
  long i = (long)blockIdx.x * blockDim.x + threadIdx.x;
  const long per = (long)EE * FSZ;
  if (i >= (long)BN * per) return;
  int b = (int)(i / per);
  long r = i - (long)b * per;
  int e = (int)(r / FSZ), n = (int)(r % FSZ);
  out[(long)b * per + (long)(e >> 5) * (FSZ * 32) + n * 32 + (e & 31)] = in[i];
}

// build bf16 padded-vertex rows (K=32) and gathered f2 voxel features (K=32)
__global__ void prep_kernel(const float* __restrict__ verts, const float* __restrict__ f2,
                            const float* __restrict__ mm,
                            unsigned short* __restrict__ vpad,
                            unsigned short* __restrict__ f2e) {
  int i = blockIdx.x * blockDim.x + threadIdx.x;   // node id over B*N
  if (i >= BN * NN) return;
  int b = i / NN;
  float vmin = mm[0], vmax = mm[1];
  float inv = 1.f / (vmax - vmin);
  const float* v = verts + (long)i * 3;
  int vi[3];
#pragma unroll
  for (int d = 0; d < 3; ++d) {
    float nrm = (v[d] - vmin) * inv;
    int q = (int)(nrm * 6.f);                       // trunc toward 0 (nrm >= 0)
    vi[d] = q < 0 ? 0 : (q > DV - 1 ? DV - 1 : q);
  }
  int lin = (vi[0] * DV + vi[1]) * DV + vi[2];
  unsigned short* vp = vpad + (long)i * 32;
  vp[0] = f2bf(v[0]); vp[1] = f2bf(v[1]); vp[2] = f2bf(v[2]);
#pragma unroll
  for (int d = 3; d < 32; ++d) vp[d] = 0;
  const float* fb = f2 + (long)b * CHN * VOX + lin;
  unsigned short* fo = f2e + (long)i * 32;
#pragma unroll
  for (int c = 0; c < CHN; ++c) fo[c] = f2bf(fb[(long)c * VOX]);
}

// final 256 -> 2 projection, writes transposed [B, 2, E]; one wave per row
__global__ __launch_bounds__(256) void final_kernel(const unsigned short* __restrict__ h,
                                                    const float* __restrict__ W,
                                                    const float* __restrict__ bias,
                                                    float* __restrict__ out) {
  int wave = threadIdx.x >> 5, lane = threadIdx.x & 31;
  long row = (long)blockIdx.x * 8 + wave;
  if (row >= (long)BN * EE) return;
  const unsigned short* hr = h + row * FSZ + lane * 8;
  float p0 = 0.f, p1 = 0.f;
#pragma unroll
  for (int j = 0; j < 8; ++j) {
    float x = bf2f(hr[j]);
    int k = lane * 8 + j;
    p0 += x * W[k * 2 + 0];
    p1 += x * W[k * 2 + 1];
  }
#pragma unroll
  for (int o = 16; o > 0; o >>= 1) {
    p0 += __shfl_down(p0, o, 32);
    p1 += __shfl_down(p1, o, 32);
  }
  if (lane == 0) {
    int b = (int)(row / EE), e = (int)(row % EE);
    out[(long)b * 2 * EE + e]      = p0 + bias[0];
    out[(long)b * 2 * EE + EE + e] = p1 + bias[1];
  }
}

// ---------------- host orchestration ----------------
extern "C" void kernel_launch(void* const* d_in, const int* in_sizes, int n_in,
                              void* d_out, int out_size, void* d_ws, size_t ws_size,
                              hipStream_t stream) {
  // inputs (pytree flatten: dict keys sorted; (W,b) pairs in order)
  const float* verts = (const float*)d_in[0];
  const float* f2    = (const float*)d_in[1];
  const int*   v1s   = (const int*)d_in[2];
  const int*   v2s   = (const int*)d_in[3];
  const float* adj   = (const float*)d_in[4];
  // mlp_f2
  const float *Wf0 = (const float*)d_in[5],  *bf0 = (const float*)d_in[6];
  const float *Wf1 = (const float*)d_in[7],  *bf1 = (const float*)d_in[8];
  // mlp_v
  const float *Wv0 = (const float*)d_in[9],  *bv0 = (const float*)d_in[10];
  const float *Wv1 = (const float*)d_in[11], *bv1 = (const float*)d_in[12];
  // stage1
  const float *Ws10 = (const float*)d_in[13], *bs10 = (const float*)d_in[14];
  const float *Ws11 = (const float*)d_in[15], *bs11 = (const float*)d_in[16];
  const float *Ws12 = (const float*)d_in[17], *bs12 = (const float*)d_in[18];
  // stage2
  const float *Ws20 = (const float*)d_in[19], *bs20 = (const float*)d_in[20];
  const float *Ws21 = (const float*)d_in[21], *bs21 = (const float*)d_in[22];
  // stage3
  const float *Ws30 = (const float*)d_in[23], *bs30 = (const float*)d_in[24];
  const float *Ws31 = (const float*)d_in[25], *bs31 = (const float*)d_in[26];
  const float *Ws32 = (const float*)d_in[27], *bs32 = (const float*)d_in[28];

  char* ws = (char*)d_ws;
  size_t off = 0;
  auto alloc = [&](size_t bytes) -> void* {
    void* p = ws + off;
    off += (bytes + 255) & ~(size_t)255;
    return p;
  };
  float* mm   = (float*)alloc(256);
  int* g1     = (int*)alloc((size_t)BN * EE * 4);
  int* g2     = (int*)alloc((size_t)BN * EE * 4);
  unsigned short* vpad = (unsigned short*)alloc((size_t)BN * NN * 32 * 2);
  unsigned short* f2e  = (unsigned short*)alloc((size_t)BN * NN * 32 * 2);
  unsigned short* wv0  = (unsigned short*)alloc(32 * 256 * 2);
  unsigned short* wv1  = (unsigned short*)alloc(256 * 256 * 2);
  unsigned short* wf0s = (unsigned short*)alloc(32 * 256 * 2);
  unsigned short* wf1s = (unsigned short*)alloc(256 * 256 * 2);
  unsigned short* w10  = (unsigned short*)alloc(1024 * 256 * 2);
  unsigned short* w11  = (unsigned short*)alloc(256 * 256 * 2);
  unsigned short* w12  = (unsigned short*)alloc(256 * 256 * 2);
  unsigned short* w20  = (unsigned short*)alloc(256 * 256 * 2);
  unsigned short* w21  = (unsigned short*)alloc(256 * 256 * 2);
  unsigned short* w30  = (unsigned short*)alloc(768 * 256 * 2);
  unsigned short* w31  = (unsigned short*)alloc(256 * 256 * 2);
  unsigned short* tA   = (unsigned short*)alloc((size_t)BN * NN * 256 * 2);
  unsigned short* tB   = (unsigned short*)alloc((size_t)BN * NN * 256 * 2);
  unsigned short* vf   = (unsigned short*)alloc((size_t)BN * NN * 512 * 2);
  unsigned short* e1   = (unsigned short*)alloc((size_t)BN * EE * 256 * 2);
  unsigned short* e2   = (unsigned short*)alloc((size_t)BN * EE * 256 * 2);
  unsigned short* s1o  = (unsigned short*)alloc((size_t)BN * EE * 256 * 2);
  unsigned short* s1sw = (unsigned short*)alloc((size_t)BN * EE * 256 * 2);
  unsigned short* agg  = (unsigned short*)alloc((size_t)BN * NN * 256 * 2);
  unsigned short* s2o  = (unsigned short*)alloc((size_t)BN * NN * 256 * 2);
  (void)ws_size; (void)n_in; (void)in_sizes; (void)out_size;

  // ---- preprocessing ----
  minmax_kernel<<<1, 1024, 0, stream>>>(verts, BN * NN * 3, mm);
  gidx_kernel<<<(BN * EE + 255) / 256, 256, 0, stream>>>(v1s, v2s, g1, g2);
  prep_kernel<<<(BN * NN + 255) / 256, 256, 0, stream>>>(verts, f2, mm, vpad, f2e);

  auto swizw = [&](const float* W, unsigned short* o, int K, int Kpad, int N) {
    swizw_kernel<<<(Kpad * N + 255) / 256, 256, 0, stream>>>(W, o, K, Kpad, N);
  };
  swizw(Wv0, wv0, 3, 32, 256);    swizw(Wv1, wv1, 256, 256, 256);
  swizw(Wf0, wf0s, 32, 32, 256);  swizw(Wf1, wf1s, 256, 256, 256);
  swizw(Ws10, w10, 1024, 1024, 256);
  swizw(Ws11, w11, 256, 256, 256); swizw(Ws12, w12, 256, 256, 256);
  swizw(Ws20, w20, 256, 256, 256); swizw(Ws21, w21, 256, 256, 256);
  swizw(Ws30, w30, 768, 768, 256); swizw(Ws31, w31, 256, 256, 256);

  auto gemm1 = [&](const void* A, const int* gth, int lda, int K,
                   const unsigned short* Bsw, const float* bias,
                   unsigned short* Out, int ldo, int M, int act) {
    GemmP p{};
    p.seg[0] = {A, gth, 0, lda, K};
    p.nseg = 1; p.Bsw = Bsw; p.bBatch = 0; p.bias = bias;
    p.Out = Out; p.ldo = ldo; p.oBatch = 0; p.M = M; p.N = 256; p.act = act;
    gemm_bf16_t<false><<<dim3(M / 128, 2, 1), 256, 0, stream>>>(p);
  };

  // mlp_v / mlp_f2 -> vertex_features [B*N, 512]
  gemm1(vpad, nullptr, 32, 32, wv0, bv0, tA, 256, BN * NN, 1);
  gemm1(tA, nullptr, 256, 256, wv1, bv1, vf, 512, BN * NN, 1);
  gemm1(f2e, nullptr, 32, 32, wf0s, bf0, tB, 256, BN * NN, 1);
  gemm1(tB, nullptr, 256, 256, wf1s, bf1, vf + 256, 512, BN * NN, 1);

  // stage1: concat(vf[g1], vf[g2]) @ W10 -> ... -> s1o
  {
    GemmP p{};
    p.seg[0] = {vf, g1, 0, 512, 512};
    p.seg[1] = {vf, g2, 0, 512, 512};
    p.nseg = 2; p.Bsw = w10; p.bBatch = 0; p.bias = bs10;
    p.Out = e1; p.ldo = 256; p.oBatch = 0; p.M = BN * EE; p.N = 256; p.act = 1;
    gemm_bf16_t<false><<<dim3(BN * EE / 128, 2, 1), 256, 0, stream>>>(p);
  }
  gemm1(e1, nullptr, 256, 256, w11, bs11, e2, 256, BN * EE, 1);
  gemm1(e2, nullptr, 256, 256, w12, bs12, s1o, 256, BN * EE, 0);

  // node aggregation: agg[b] = adj[b] @ s1o[b]  (f32 A staged via LDS)
  swizb_kernel<<<(int)(((long)BN * EE * FSZ + 255) / 256), 256, 0, stream>>>(s1o, s1sw);
  {
    GemmP p{};
    p.seg[0] = {adj, nullptr, (long)NN * EE, EE, EE};
    p.nseg = 1; p.Bsw = s1sw; p.bBatch = (long)EE * FSZ; p.bias = nullptr;
    p.Out = agg; p.ldo = 256; p.oBatch = (long)NN * FSZ;
    p.M = NN; p.N = 256; p.act = 0;
    gemm_bf16_t<true><<<dim3(NN / 128, 2, BN), 256, 0, stream>>>(p);
  }

  // stage2
  gemm1(agg, nullptr, 256, 256, w20, bs20, tA, 256, BN * NN, 1);
  gemm1(tA, nullptr, 256, 256, w21, bs21, s2o, 256, BN * NN, 1);

  // stage3: concat(s1o, s2o[g1], s2o[g2]) @ W30 -> ELU -> @W31 -> ELU
  {
    GemmP p{};
    p.seg[0] = {s1o, nullptr, 0, 256, 256};
    p.seg[1] = {s2o, g1, 0, 256, 256};
    p.seg[2] = {s2o, g2, 0, 256, 256};
    p.nseg = 3; p.Bsw = w30; p.bBatch = 0; p.bias = bs30;
    p.Out = e1; p.ldo = 256; p.oBatch = 0; p.M = BN * EE; p.N = 256; p.act = 1;
    gemm_bf16_t<false><<<dim3(BN * EE / 128, 2, 1), 256, 0, stream>>>(p);
  }
  gemm1(e1, nullptr, 256, 256, w31, bs31, e2, 256, BN * EE, 1);

  // final 256 -> 2, transposed output [B, 2, E]
  final_kernel<<<BN * EE / 8, 256, 0, stream>>>(e2, Ws32, bs32, (float*)d_out);
}
